// FusedSparseECMoeBlock_43121471652485
// MI455X (gfx1250) — matrix operand order repaired and checked by reading, not compiled, not run on hardware
//
#include <hip/hip_runtime.h>
#include <cstdint>
#include <cstddef>

typedef __bf16 bf16_t;
typedef __attribute__((ext_vector_type(16))) __bf16 v16bf;
typedef __attribute__((ext_vector_type(8)))  __bf16 v8bf;
typedef __attribute__((ext_vector_type(8)))  float  v8f;

#define TOKENS 8192
#define NEXP   16
#define DMODEL 1024
#define FDIM   4096
#define SHDIM  2048
#define CAP    1024
#define LDKP   40   // padded LDS row stride (bf16 elems) for 32-wide K tiles

static __device__ __forceinline__ v8f zero_v8f() {
  v8f z = {0.f, 0.f, 0.f, 0.f, 0.f, 0.f, 0.f, 0.f};
  return z;
}

static __device__ __forceinline__ float silu_f(float x) {
  return x / (1.f + __expf(-x));
}

static __device__ __forceinline__ unsigned pack_bf16x2(float x, float y) {
  union { bf16_t b[2]; unsigned u; } cv;
  cv.b[0] = (bf16_t)x; cv.b[1] = (bf16_t)y;
  return cv.u;
}

// convert 8 contiguous f32 -> 8 bf16, store as one 16B LDS write
static __device__ __forceinline__ void cvt_store8(bf16_t* dst, const float* src) {
  float4 a = *(const float4*)src;
  float4 b = *(const float4*)(src + 4);
  uint4 v;
  v.x = pack_bf16x2(a.x, a.y);
  v.y = pack_bf16x2(a.z, a.w);
  v.z = pack_bf16x2(b.x, b.y);
  v.w = pack_bf16x2(b.z, b.w);
  *(uint4*)dst = v;
}

// A fragment (16x32 bf16): lane<16 -> row=lane, k in {0..7,16..23};
// lane>=16 -> row=lane-16, k in {8..15,24..31}   (ISA 7.12.2)
static __device__ __forceinline__ v16bf frag_a(const bf16_t* tile) {
  const int lane = threadIdx.x & 31;
  const int r    = lane & 15;
  const int koff = (lane & 16) ? 8 : 0;
  const bf16_t* p = tile + r * LDKP + koff;
  v8bf lo = *(const v8bf*)p;
  v8bf hi = *(const v8bf*)(p + 16);
  return __builtin_shufflevector(lo, hi, 0,1,2,3,4,5,6,7,8,9,10,11,12,13,14,15);
}

// B fragment (32x16 bf16) staged as [N][K]: lane<16 -> n=lane, k in {0..7,16..23};
// lane>=16 -> n=lane-16, k in {8..15,24..31}
static __device__ __forceinline__ v16bf frag_b(const bf16_t* tile) {
  const int lane = threadIdx.x & 31;
  const int n    = lane & 15;
  const int koff = (lane & 16) ? 8 : 0;
  const bf16_t* p = tile + n * LDKP + koff;
  v8bf lo = *(const v8bf*)p;
  v8bf hi = *(const v8bf*)(p + 16);
  return __builtin_shufflevector(lo, hi, 0,1,2,3,4,5,6,7,8,9,10,11,12,13,14,15);
}

static __device__ __forceinline__ v8f wmma_bf16(v16bf a, v16bf b, v8f c) {
  return __builtin_amdgcn_wmma_f32_16x16x32_bf16(false, a, false, b, (short)0, c,
                                                 false, false);
}

// ---------------------------------------------------------------- router gate
__global__ void gate_kernel(const float* __restrict__ x,
                            const float* __restrict__ gw,
                            float* __restrict__ scores) {
  const int tid = threadIdx.x;
  const int t = blockIdx.x * 16 + (tid >> 4);
  const int e = tid & 15;
  const float* xr = x  + (size_t)t * DMODEL;
  const float* wr = gw + (size_t)e * DMODEL;
  float acc = 0.f;
#pragma unroll 8
  for (int k = 0; k < DMODEL; k += 4) {
    float4 xv = *(const float4*)(xr + k);
    float4 wv = *(const float4*)(wr + k);
    acc = fmaf(xv.x, wv.x, acc);
    acc = fmaf(xv.y, wv.y, acc);
    acc = fmaf(xv.z, wv.z, acc);
    acc = fmaf(xv.w, wv.w, acc);
  }
  scores[(size_t)t * NEXP + e] = 1.f / (1.f + __expf(-acc));
}

// --------------------------------------------- per-expert top-CAP radix select
__global__ void topk_kernel(const float* __restrict__ scores,
                            int*  __restrict__ topk_ids,
                            float* __restrict__ topk_sc) {
  const int e   = blockIdx.x;
  const int tid = threadIdx.x;
  __shared__ uint32_t keys[TOKENS];   // sigmoid>0 => uint bits order == float order
  __shared__ unsigned scnt;

  for (int t = tid; t < TOKENS; t += blockDim.x)
    keys[t] = __float_as_uint(scores[(size_t)t * NEXP + e]);
  __syncthreads();

  uint32_t thresh = 0;
  for (int bit = 31; bit >= 0; --bit) {
    const uint32_t cand = thresh | (1u << bit);
    if (tid == 0) scnt = 0;
    __syncthreads();
    unsigned local = 0;
    for (int t = tid; t < TOKENS; t += blockDim.x)
      local += (keys[t] >= cand);
    if (local) atomicAdd(&scnt, local);
    __syncthreads();
    if (scnt >= CAP) thresh = cand;
    __syncthreads();
  }

  if (tid == 0) scnt = 0;
  __syncthreads();
  for (int t = tid; t < TOKENS; t += blockDim.x) {
    if (keys[t] > thresh) {
      unsigned pos = atomicAdd(&scnt, 1u);
      topk_ids[(size_t)e * CAP + pos] = t;
      topk_sc [(size_t)e * CAP + pos] = __uint_as_float(keys[t]);
    }
  }
  __syncthreads();
  for (int t = tid; t < TOKENS; t += blockDim.x) {
    if (keys[t] == thresh) {
      unsigned pos = atomicAdd(&scnt, 1u);
      if (pos < CAP) {
        topk_ids[(size_t)e * CAP + pos] = t;
        topk_sc [(size_t)e * CAP + pos] = __uint_as_float(keys[t]);
      }
    }
  }
}

// ------------------------------------------------ fused up/gate GEMM + SiLU*up
// Block tile 128(M) x 64(N); 8 waves in 4(M) x 2(N) grid; 32x32 per wave.
// Per wave: 2x2 C tiles per GEMM, two GEMMs (up & gate) = 8 v8f accumulators.
template <bool EXPERT>
__global__ void __launch_bounds__(256, 1)
glu_ff1_kernel(const float* __restrict__ X,
               const float* __restrict__ Wu,
               const float* __restrict__ Wg,
               const float* __restrict__ Bu,
               const float* __restrict__ Bg,
               const int*   __restrict__ ids,
               bf16_t* __restrict__ H,
               int M, int N, int K) {
  const int tid  = threadIdx.x;
  const int lane = tid & 31;
  const int w    = tid >> 5;
  const int wm   = (w & 3) * 32;   // 0,32,64,96
  const int wn   = (w >> 2) * 32;  // 0,32
  const int m0   = blockIdx.x * 128;
  const int n0   = blockIdx.y * 64;
  const size_t e = EXPERT ? (size_t)blockIdx.z : 0;

  __shared__ __align__(16) bf16_t As[128 * LDKP];
  __shared__ __align__(16) bf16_t Us[64 * LDKP];
  __shared__ __align__(16) bf16_t Gs[64 * LDKP];
  __shared__ int sIds[128];

  const float* Wu_e = Wu + e * (size_t)N * K + (size_t)n0 * K;
  const float* Wg_e = Wg + e * (size_t)N * K + (size_t)n0 * K;

  if (EXPERT) {
    if (tid < 128) sIds[tid] = ids[e * M + m0 + tid];
  }

  v8f accU[2][2], accG[2][2];
#pragma unroll
  for (int i = 0; i < 2; ++i)
#pragma unroll
    for (int j = 0; j < 2; ++j) { accU[i][j] = zero_v8f(); accG[i][j] = zero_v8f(); }

  for (int k0 = 0; k0 < K; k0 += 32) {
    __syncthreads();
    // stage A (128x32): 512 8-elem chunks over 256 threads
    for (int idx = tid; idx < 512; idx += 256) {
      const int r = idx >> 2;
      const int c = (idx & 3) << 3;
      const float* src = EXPERT ? (X + (size_t)sIds[r] * K + k0 + c)
                                : (X + (size_t)(m0 + r) * K + k0 + c);
      cvt_store8(&As[r * LDKP + c], src);
    }
    // stage Wu/Wg (64x32 each): 256 chunks each, one per thread
    {
      const int r = tid >> 2;
      const int c = (tid & 3) << 3;
      cvt_store8(&Us[r * LDKP + c], Wu_e + (size_t)r * K + k0 + c);
      cvt_store8(&Gs[r * LDKP + c], Wg_e + (size_t)r * K + k0 + c);
    }
    __syncthreads();

    v16bf af[2], uf[2], gf[2];
#pragma unroll
    for (int i = 0; i < 2; ++i) af[i] = frag_a(&As[(wm + i * 16) * LDKP]);
#pragma unroll
    for (int j = 0; j < 2; ++j) {
      uf[j] = frag_b(&Us[(wn + j * 16) * LDKP]);
      gf[j] = frag_b(&Gs[(wn + j * 16) * LDKP]);
    }
#pragma unroll
    for (int i = 0; i < 2; ++i)
#pragma unroll
      for (int j = 0; j < 2; ++j) {
        accU[i][j] = wmma_bf16(af[i], uf[j], accU[i][j]);
        accG[i][j] = wmma_bf16(af[i], gf[j], accG[i][j]);
      }
  }

  // epilogue: h = silu(g + bg) * (u + bu) -> bf16
  const int lmbase = (lane & 16) ? 8 : 0;
  const int ncol   = lane & 15;
  bf16_t* He = H + (EXPERT ? e * (size_t)M * N : 0);
#pragma unroll
  for (int j = 0; j < 2; ++j) {
    const int ng = n0 + wn + j * 16 + ncol;
    const float bu = EXPERT ? Bu[e * (size_t)N + ng] : 0.f;
    const float bg = EXPERT ? Bg[e * (size_t)N + ng] : 0.f;
#pragma unroll
    for (int i = 0; i < 2; ++i) {
#pragma unroll
      for (int r = 0; r < 8; ++r) {
        const int mg = m0 + wm + i * 16 + lmbase + r;
        const float u = accU[i][j][r] + bu;
        const float g = accG[i][j][r] + bg;
        He[(size_t)mg * N + ng] = (bf16_t)(silu_f(g) * u);
      }
    }
  }
}

// ------------------------------------------------------- down GEMM (+ scatter)
// Block tile 128x128; 8 waves in 2(M) x 4(N) grid; 64x32 per wave (8 acc tiles).
template <bool SCATTER>
__global__ void __launch_bounds__(256, 1)
ff2_kernel(const bf16_t* __restrict__ H,
           const float* __restrict__ W,
           const float* __restrict__ Bias,
           const int*   __restrict__ ids,
           const float* __restrict__ gsc,
           float* __restrict__ out,
           int M, int N, int K) {
  const int tid  = threadIdx.x;
  const int lane = tid & 31;
  const int w    = tid >> 5;
  const int wm   = (w & 1) * 64;
  const int wn   = (w >> 1) * 32;
  const int m0   = blockIdx.x * 128;
  const int n0   = blockIdx.y * 128;
  const size_t e = SCATTER ? (size_t)blockIdx.z : 0;

  __shared__ __align__(16) bf16_t As[128 * LDKP];
  __shared__ __align__(16) bf16_t Bs[128 * LDKP];
  __shared__ int   sIds[128];
  __shared__ float sSc[128];

  const bf16_t* He = H + (SCATTER ? e * (size_t)M * K : 0) + (size_t)m0 * K;
  const float*  We = W + e * (size_t)N * K + (size_t)n0 * K;

  if (SCATTER) {
    if (tid < 128) {
      sIds[tid] = ids[e * M + m0 + tid];
      sSc [tid] = gsc[e * M + m0 + tid];
    }
  }

  v8f acc[4][2];
#pragma unroll
  for (int i = 0; i < 4; ++i)
#pragma unroll
    for (int j = 0; j < 2; ++j) acc[i][j] = zero_v8f();

  for (int k0 = 0; k0 < K; k0 += 32) {
    __syncthreads();
    for (int idx = tid; idx < 512; idx += 256) {
      const int r = idx >> 2;
      const int c = (idx & 3) << 3;
      *(uint4*)&As[r * LDKP + c] = *(const uint4*)(He + (size_t)r * K + k0 + c);
      cvt_store8(&Bs[r * LDKP + c], We + (size_t)r * K + k0 + c);
    }
    __syncthreads();

    v16bf af[4], bf[2];
#pragma unroll
    for (int i = 0; i < 4; ++i) af[i] = frag_a(&As[(wm + i * 16) * LDKP]);
#pragma unroll
    for (int j = 0; j < 2; ++j) bf[j] = frag_b(&Bs[(wn + j * 16) * LDKP]);
#pragma unroll
    for (int i = 0; i < 4; ++i)
#pragma unroll
      for (int j = 0; j < 2; ++j)
        acc[i][j] = wmma_bf16(af[i], bf[j], acc[i][j]);
  }

  const int lmbase = (lane & 16) ? 8 : 0;
  const int ncol   = lane & 15;
#pragma unroll
  for (int j = 0; j < 2; ++j) {
    const int ng = n0 + wn + j * 16 + ncol;
    const float bias = SCATTER ? Bias[e * (size_t)N + ng] : 0.f;
#pragma unroll
    for (int i = 0; i < 4; ++i) {
#pragma unroll
      for (int r = 0; r < 8; ++r) {
        const int ml = wm + i * 16 + lmbase + r;
        if (SCATTER) {
          const float v = (acc[i][j][r] + bias) * sSc[ml];
          atomicAdd(out + (size_t)sIds[ml] * N + ng, v);
        } else {
          out[(size_t)(m0 + ml) * N + ng] = acc[i][j][r];
        }
      }
    }
  }
}

// ------------------------------------------------------------------- launcher
extern "C" void kernel_launch(void* const* d_in, const int* in_sizes, int n_in,
                              void* d_out, int out_size, void* d_ws, size_t ws_size,
                              hipStream_t stream) {
  (void)in_sizes; (void)n_in; (void)out_size; (void)ws_size;
  const float* x        = (const float*)d_in[0];
  const float* gate_w   = (const float*)d_in[1];
  const float* up_w     = (const float*)d_in[2];
  const float* up_b     = (const float*)d_in[3];
  const float* gw       = (const float*)d_in[4];
  const float* gb       = (const float*)d_in[5];
  const float* down_w   = (const float*)d_in[6];
  const float* down_b   = (const float*)d_in[7];
  const float* sh_gate  = (const float*)d_in[8];
  const float* sh_up    = (const float*)d_in[9];
  const float* sh_down  = (const float*)d_in[10];
  float* out = (float*)d_out;

  char* ws = (char*)d_ws;
  float*  scores   = (float*)ws;                               // 8192*16*4 = 512KB
  int*    topk_ids = (int*)  (ws + 524288);                    // 16*1024*4 = 64KB
  float*  topk_sc  = (float*)(ws + 524288 + 65536);            // 64KB
  bf16_t* Hexp     = (bf16_t*)(ws + 655360);                   // 16*1024*4096*2 = 128MB
  bf16_t* Hsh      = (bf16_t*)(ws + 655360 + (size_t)NEXP * CAP * FDIM * 2);

  // 1) router scores
  gate_kernel<<<dim3(TOKENS / 16), 256, 0, stream>>>(x, gate_w, scores);
  // 2) expert-choice top-k
  topk_kernel<<<dim3(NEXP), 256, 0, stream>>>(scores, topk_ids, topk_sc);
  // 3) shared GLU up/gate (no bias, no gather)
  glu_ff1_kernel<false><<<dim3(TOKENS / 128, SHDIM / 64, 1), 256, 0, stream>>>(
      x, sh_up, sh_gate, nullptr, nullptr, nullptr, Hsh, TOKENS, SHDIM, DMODEL);
  // 4) shared down -> writes out (must precede scatter-adds)
  ff2_kernel<false><<<dim3(TOKENS / 128, DMODEL / 128, 1), 256, 0, stream>>>(
      Hsh, sh_down, nullptr, nullptr, nullptr, out, TOKENS, DMODEL, SHDIM);
  // 5) expert GLU up/gate with token gather + biases
  glu_ff1_kernel<true><<<dim3(CAP / 128, FDIM / 64, NEXP), 256, 0, stream>>>(
      x, up_w, gw, up_b, gb, topk_ids, Hexp, CAP, FDIM, DMODEL);
  // 6) expert down + weighted atomic scatter-add into out
  ff2_kernel<true><<<dim3(CAP / 128, DMODEL / 128, NEXP), 256, 0, stream>>>(
      Hexp, down_w, down_b, topk_ids, topk_sc, out, CAP, DMODEL, FDIM);
}
